// ConvBlock_26809185861899
// MI455X (gfx1250) — matrix operand order repaired and checked by reading, not compiled, not run on hardware
//
#include <hip/hip_runtime.h>
#include <hip/hip_bf16.h>

typedef __attribute__((ext_vector_type(16))) _Float16 v16h;
typedef __attribute__((ext_vector_type(8)))  _Float16 v8h;
typedef __attribute__((ext_vector_type(8)))  float    v8f;

#define BATCH   4
#define NPTS    16384
#define NPOINT  1024
#define NSAMPLE 32
#define CIN     64
#define CPAD    96      /* 3 (xyz) + 64 (feat) padded to 96 for K-tiling */
#define C0      64
#define C1      128
#define NCOLS   (BATCH * NPOINT * NSAMPLE)   /* 131072 */
#define NBLK    (BATCH * NPOINT)             /* 4096 */
#define R2      0.04f
#define BN_EPS  1e-5f

// ---------------- workspace layout (bytes) ----------------
#define WS_FPS    0                    /* int[4096]                        */
#define WS_SCALE  16384                /* scale0[64] shift0[64] scale1[128] shift1[128] */
#define WS_W0H    17920                /* f16 [64][96]                     */
#define WS_W1H    30208                /* f16 [128][64]                    */
#define WS_PART0  46592                /* f32 [4096][128] (sum|sumsq)      */
#define WS_PART1  2143744              /* f32 [4096][256]                  */
#define WS_XG     6338048              /* f16 [131072][96] column-major    */

// ================= prep: weights -> padded f16 =================
__global__ void prep_kernel(const float* __restrict__ W0, const float* __restrict__ W1,
                            _Float16* __restrict__ W0h, _Float16* __restrict__ W1h) {
    int tid = threadIdx.x + blockIdx.x * blockDim.x;
    int stride = blockDim.x * gridDim.x;
    for (int i = tid; i < C0 * CPAD; i += stride) {
        int o = i / CPAD, k = i % CPAD;
        W0h[i] = (k < 67) ? (_Float16)W0[o * 67 + k] : (_Float16)0.0f;
    }
    for (int i = tid; i < C1 * C0; i += stride)
        W1h[i] = (_Float16)W1[i];
}

// ================= furthest point sampling =================
// one 1024-thread block per batch; 16 points/thread held in registers
__global__ void fps_kernel(const float* __restrict__ xyz, int* __restrict__ fps_idx) {
    const int b = blockIdx.x;
    const int t = threadIdx.x;
    const int lane = t & 31, warp = t >> 5;
    const float* base = xyz + (size_t)b * NPTS * 3;
    float px[16], py[16], pz[16], mind[16];
#pragma unroll
    for (int j = 0; j < 16; ++j) {
        int p = t + j * 1024;
        px[j] = base[p * 3 + 0];
        py[j] = base[p * 3 + 1];
        pz[j] = base[p * 3 + 2];
        mind[j] = 1e10f;
    }
    __shared__ float s_val[32];
    __shared__ int   s_idx[32];
    __shared__ int   s_far;
    int far = 0;
    for (int i = 0; i < NPOINT; ++i) {
        if (t == 0) fps_idx[b * NPOINT + i] = far;
        float cx = base[far * 3 + 0], cy = base[far * 3 + 1], cz = base[far * 3 + 2];
        float best = -1.0f; int bi = 0x7fffffff;
#pragma unroll
        for (int j = 0; j < 16; ++j) {
            float dx = px[j] - cx, dy = py[j] - cy, dz = pz[j] - cz;
            float d = dx * dx + dy * dy + dz * dz;
            float m = fminf(mind[j], d);
            mind[j] = m;
            int p = t + j * 1024;
            if (m > best || (m == best && p < bi)) { best = m; bi = p; }
        }
#pragma unroll
        for (int off = 16; off >= 1; off >>= 1) {
            float ov = __shfl_xor(best, off, 32);
            int   oi = __shfl_xor(bi,   off, 32);
            if (ov > best || (ov == best && oi < bi)) { best = ov; bi = oi; }
        }
        if (lane == 0) { s_val[warp] = best; s_idx[warp] = bi; }
        __syncthreads();
        if (t < 32) {
            best = s_val[t]; bi = s_idx[t];
#pragma unroll
            for (int off = 16; off >= 1; off >>= 1) {
                float ov = __shfl_xor(best, off, 32);
                int   oi = __shfl_xor(bi,   off, 32);
                if (ov > best || (ov == best && oi < bi)) { best = ov; bi = oi; }
            }
            if (t == 0) s_far = bi;
        }
        __syncthreads();
        far = s_far;
        __syncthreads();
    }
}

// ================= ball query + gather -> xg (f16, col-major [col][96]) =================
__global__ void group_kernel(const float* __restrict__ xyz, const float* __restrict__ feat,
                             const int* __restrict__ fps_idx, _Float16* __restrict__ xg,
                             float* __restrict__ out_xyz) {
    const int bs = blockIdx.x;            // 0..4095
    const int b = bs >> 10;
    const int lane = threadIdx.x;         // 0..31
    const float* base = xyz + (size_t)b * NPTS * 3;
    int far = fps_idx[bs];
    float cx = base[far * 3 + 0], cy = base[far * 3 + 1], cz = base[far * 3 + 2];
    if (lane == 0) {
        out_xyz[bs * 3 + 0] = cx;
        out_xyz[bs * 3 + 1] = cy;
        out_xyz[bs * 3 + 2] = cz;
    }
    __shared__ int s_nb[32];
    int cnt = 0;
    for (int b0 = 0; b0 < NPTS; b0 += 32) {
        int p = b0 + lane;
        float dx = base[p * 3 + 0] - cx;
        float dy = base[p * 3 + 1] - cy;
        float dz = base[p * 3 + 2] - cz;
        bool pred = (dx * dx + dy * dy + dz * dz) < R2;
        unsigned mask = (unsigned)__ballot(pred);
        int tot = __popc(mask);
        int pos = __popc(mask & ((1u << lane) - 1u));
        if (pred && (cnt + pos) < NSAMPLE) s_nb[cnt + pos] = p;
        cnt += tot;
        if (cnt >= NSAMPLE) break;
    }
    __syncthreads();
    int pt = s_nb[(lane < cnt) ? lane : 0];   // pad with first neighbor
    // build one 96-channel column in registers, store as 6x 32B vectors
    v16h ob[6];
    float gx = base[pt * 3 + 0] - cx;
    float gy = base[pt * 3 + 1] - cy;
    float gz = base[pt * 3 + 2] - cz;
    ob[0][0] = (_Float16)gx;
    ob[0][1] = (_Float16)gy;
    ob[0][2] = (_Float16)gz;
#pragma unroll
    for (int ci = 0; ci < CIN; ++ci) {
        float f = feat[((size_t)b * CIN + ci) * NPTS + pt];
        int e = ci + 3;
        ob[e >> 4][e & 15] = (_Float16)f;
    }
#pragma unroll
    for (int e = 67; e < CPAD; ++e) ob[e >> 4][e & 15] = (_Float16)0.0f;
    v16h* dst = (v16h*)(xg + ((size_t)bs * NSAMPLE + lane) * CPAD);
#pragma unroll
    for (int i = 0; i < 6; ++i) dst[i] = ob[i];
}

// ================= WMMA helpers =================
__device__ __forceinline__ v16h load_a_tile(const _Float16* p) {
    v8h lo = *(const v8h*)(p);
    v8h hi = *(const v8h*)(p + 8);
    return __builtin_shufflevector(lo, hi, 0, 1, 2, 3, 4, 5, 6, 7,
                                   8, 9, 10, 11, 12, 13, 14, 15);
}

// conv0: (64x96)*(96x32) -> acc[4 mtiles][2 ntiles], one wave per center
__device__ __forceinline__ void conv0_tiles(const _Float16* __restrict__ xg,
                                            const _Float16* __restrict__ W0h,
                                            int bs, v8f acc[4][2]) {
    const int lane = threadIdx.x & 31;
    const int half = lane >> 4, l16 = lane & 15;
    const _Float16* colb = xg + (size_t)bs * NSAMPLE * CPAD;
    v16h Bt[2][3];
#pragma unroll
    for (int nt = 0; nt < 2; ++nt) {
        const _Float16* cp = colb + (nt * 16 + l16) * CPAD + half * 16;
#pragma unroll
        for (int kt = 0; kt < 3; ++kt)
            Bt[nt][kt] = *(const v16h*)(cp + kt * 32);
    }
#pragma unroll
    for (int mt = 0; mt < 4; ++mt) {
        const _Float16* wrow = W0h + (mt * 16 + l16) * CPAD + half * 16;
        v16h A[3];
#pragma unroll
        for (int kt = 0; kt < 3; ++kt) A[kt] = load_a_tile(wrow + kt * 32);
#pragma unroll
        for (int nt = 0; nt < 2; ++nt) {
            v8f c = {};
#pragma unroll
            for (int kt = 0; kt < 3; ++kt)
                c = __builtin_amdgcn_wmma_f32_16x16x32_f16(
                        false, A[kt], false, Bt[nt][kt], (short)0, c, false, false);
            acc[mt][nt] = c;
        }
    }
}

// BN0 + ReLU, write h[32 cols][64 ch] (col-major) into LDS
__device__ __forceinline__ void bn0_to_lds(const v8f acc[4][2],
                                           const float* __restrict__ scale0,
                                           const float* __restrict__ shift0,
                                           _Float16* h) {
    const int lane = threadIdx.x & 31;
    const int half = lane >> 4, l16 = lane & 15;
#pragma unroll
    for (int mt = 0; mt < 4; ++mt)
#pragma unroll
        for (int v = 0; v < 8; ++v) {
            int ch = mt * 16 + half * 8 + v;
            float sc = scale0[ch], sh = shift0[ch];
#pragma unroll
            for (int nt = 0; nt < 2; ++nt) {
                int col = nt * 16 + l16;
                float x = fmaxf(fmaf(acc[mt][nt][v], sc, sh), 0.0f);
                h[col * C0 + ch] = (_Float16)x;
            }
        }
}

__device__ __forceinline__ void load_h_tiles(const _Float16* h, v16h Bt[2][2]) {
    const int lane = threadIdx.x & 31;
    const int half = lane >> 4, l16 = lane & 15;
#pragma unroll
    for (int nt = 0; nt < 2; ++nt)
#pragma unroll
        for (int kt = 0; kt < 2; ++kt)
            Bt[nt][kt] = *(const v16h*)(h + (nt * 16 + l16) * C0 + kt * 32 + half * 16);
}

// conv1: one m-tile of (128x64)*(64x32)
__device__ __forceinline__ void conv1_mtile(const _Float16* __restrict__ W1h,
                                            const v16h Bt[2][2], int mt, v8f out[2]) {
    const int lane = threadIdx.x & 31;
    const int half = lane >> 4, l16 = lane & 15;
    const _Float16* wrow = W1h + (mt * 16 + l16) * C0 + half * 16;
    v16h A[2];
#pragma unroll
    for (int kt = 0; kt < 2; ++kt) A[kt] = load_a_tile(wrow + kt * 32);
#pragma unroll
    for (int nt = 0; nt < 2; ++nt) {
        v8f c = {};
#pragma unroll
        for (int kt = 0; kt < 2; ++kt)
            c = __builtin_amdgcn_wmma_f32_16x16x32_f16(
                    false, A[kt], false, Bt[nt][kt], (short)0, c, false, false);
        out[nt] = c;
    }
}

// deterministic per-block partial stats (sum | sumsq) for one 16-row m-tile
__device__ __forceinline__ void accum_part(const v8f& c0, const v8f& c1, int m0,
                                           float* __restrict__ part_blk, int C) {
    const int lane = threadIdx.x & 31;
    const int half = lane >> 4, l16 = lane & 15;
#pragma unroll
    for (int v = 0; v < 8; ++v) {
        float s = c0[v] + c1[v];
        float q = c0[v] * c0[v] + c1[v] * c1[v];
#pragma unroll
        for (int off = 1; off < 16; off <<= 1) {
            s += __shfl_xor(s, off, 32);
            q += __shfl_xor(q, off, 32);
        }
        if (l16 == 0) {
            int ch = m0 + half * 8 + v;
            part_blk[ch]     = s;
            part_blk[C + ch] = q;
        }
    }
}

// ================= pass kernels =================
__global__ void conv0_stats_kernel(const _Float16* __restrict__ xg,
                                   const _Float16* __restrict__ W0h,
                                   float* __restrict__ part0) {
    int bs = blockIdx.x;
    v8f a0[4][2];
    conv0_tiles(xg, W0h, bs, a0);
    float* pb = part0 + (size_t)bs * (2 * C0);
#pragma unroll
    for (int mt = 0; mt < 4; ++mt)
        accum_part(a0[mt][0], a0[mt][1], mt * 16, pb, C0);
}

__global__ void finalize_kernel(const float* __restrict__ part, int C,
                                const float* __restrict__ g, const float* __restrict__ bb,
                                float* __restrict__ scale, float* __restrict__ shift) {
    int ch = threadIdx.x;
    if (ch >= C) return;
    float s = 0.0f, q = 0.0f;
    int stride = 2 * C;
    for (int blk = 0; blk < NBLK; ++blk) {
        s += part[(size_t)blk * stride + ch];
        q += part[(size_t)blk * stride + C + ch];
    }
    const float inv_n = 1.0f / (float)NCOLS;
    float mean = s * inv_n;
    float var  = q * inv_n - mean * mean;
    float sc = g[ch] * rsqrtf(var + BN_EPS);
    scale[ch] = sc;
    shift[ch] = bb[ch] - mean * sc;
}

__global__ void conv1_stats_kernel(const _Float16* __restrict__ xg,
                                   const _Float16* __restrict__ W0h,
                                   const _Float16* __restrict__ W1h,
                                   const float* __restrict__ scale0,
                                   const float* __restrict__ shift0,
                                   float* __restrict__ part1) {
    __shared__ __align__(32) _Float16 h[NSAMPLE * C0];
    int bs = blockIdx.x;
    v8f a0[4][2];
    conv0_tiles(xg, W0h, bs, a0);
    bn0_to_lds(a0, scale0, shift0, h);
    __syncthreads();
    v16h Bt[2][2];
    load_h_tiles(h, Bt);
    float* pb = part1 + (size_t)bs * (2 * C1);
#pragma unroll
    for (int mt = 0; mt < 8; ++mt) {
        v8f c[2];
        conv1_mtile(W1h, Bt, mt, c);
        accum_part(c[0], c[1], mt * 16, pb, C1);
    }
}

__global__ void final_kernel(const _Float16* __restrict__ xg,
                             const _Float16* __restrict__ W0h,
                             const _Float16* __restrict__ W1h,
                             const float* __restrict__ scale0,
                             const float* __restrict__ shift0,
                             const float* __restrict__ scale1,
                             const float* __restrict__ shift1,
                             float* __restrict__ outf) {
    __shared__ __align__(32) _Float16 h[NSAMPLE * C0];
    int bs = blockIdx.x;
    int b = bs >> 10, s = bs & 1023;
    const int lane = threadIdx.x & 31;
    const int half = lane >> 4, l16 = lane & 15;
    v8f a0[4][2];
    conv0_tiles(xg, W0h, bs, a0);
    bn0_to_lds(a0, scale0, shift0, h);
    __syncthreads();
    v16h Bt[2][2];
    load_h_tiles(h, Bt);
#pragma unroll
    for (int mt = 0; mt < 8; ++mt) {
        v8f c[2];
        conv1_mtile(W1h, Bt, mt, c);
#pragma unroll
        for (int v = 0; v < 8; ++v) {
            int ch = mt * 16 + half * 8 + v;
            float sc = scale1[ch], sh = shift1[ch];
            float r0 = fmaxf(fmaf(c[0][v], sc, sh), 0.0f);
            float r1 = fmaxf(fmaf(c[1][v], sc, sh), 0.0f);
            float mx = fmaxf(r0, r1);
#pragma unroll
            for (int off = 1; off < 16; off <<= 1)
                mx = fmaxf(mx, __shfl_xor(mx, off, 32));
            if (l16 == 0)
                outf[((size_t)(b * C1 + ch)) * NPOINT + s] = mx;
        }
    }
}

// ================= launch =================
extern "C" void kernel_launch(void* const* d_in, const int* in_sizes, int n_in,
                              void* d_out, int out_size, void* d_ws, size_t ws_size,
                              hipStream_t stream) {
    const float* xyz  = (const float*)d_in[0];
    const float* feat = (const float*)d_in[1];
    const float* W0   = (const float*)d_in[2];
    const float* g0   = (const float*)d_in[3];
    const float* b0   = (const float*)d_in[4];
    const float* W1   = (const float*)d_in[5];
    const float* g1   = (const float*)d_in[6];
    const float* b1   = (const float*)d_in[7];

    char* ws = (char*)d_ws;
    int*      fps_idx = (int*)(ws + WS_FPS);
    float*    scale0  = (float*)(ws + WS_SCALE);
    float*    shift0  = scale0 + 64;
    float*    scale1  = scale0 + 128;
    float*    shift1  = scale0 + 256;
    _Float16* W0h     = (_Float16*)(ws + WS_W0H);
    _Float16* W1h     = (_Float16*)(ws + WS_W1H);
    float*    part0   = (float*)(ws + WS_PART0);
    float*    part1   = (float*)(ws + WS_PART1);
    _Float16* xg      = (_Float16*)(ws + WS_XG);

    float* out_xyz  = (float*)d_out;                    // [B,S,3]  = 12288
    float* out_feat = (float*)d_out + BATCH * NPOINT * 3;

    prep_kernel<<<32, 256, 0, stream>>>(W0, W1, W0h, W1h);
    fps_kernel<<<BATCH, 1024, 0, stream>>>(xyz, fps_idx);
    group_kernel<<<NBLK, 32, 0, stream>>>(xyz, feat, fps_idx, xg, out_xyz);
    conv0_stats_kernel<<<NBLK, 32, 0, stream>>>(xg, W0h, part0);
    finalize_kernel<<<1, 64, 0, stream>>>(part0, C0, g0, b0, scale0, shift0);
    conv1_stats_kernel<<<NBLK, 32, 0, stream>>>(xg, W0h, W1h, scale0, shift0, part1);
    finalize_kernel<<<1, 128, 0, stream>>>(part1, C1, g1, b1, scale1, shift1);
    final_kernel<<<NBLK, 32, 0, stream>>>(xg, W0h, W1h, scale0, shift0,
                                          scale1, shift1, out_feat);
}
